// Attention_24867860644486
// MI455X (gfx1250) — compile-verified, matrix operand order checked by reading, not run on hardware
//
#include <hip/hip_runtime.h>
#include <math.h>

#define Bb 4
#define Nn 2048
#define Cc 256
#define Hh 8
#define SCALE 0.0625f   // 256^-0.5

// Use the CDNA5 async global->LDS DMA path (ASYNCcnt) via inline asm.
#define USE_ASYNC_LDS 1

typedef __attribute__((ext_vector_type(16))) __bf16 v16bf;
typedef __attribute__((ext_vector_type(8)))  float  v8f;

union FragU { uint4 q[2]; v16bf v; };

__device__ __forceinline__ unsigned short f2bf(float f) {
    unsigned int u = __float_as_uint(f);
    u += 0x7FFFu + ((u >> 16) & 1u);           // round-to-nearest-even
    return (unsigned short)(u >> 16);
}

// Universal 16-bit WMMA fragment loader (A 16x32 and B 32x16 layouts coincide):
// lane L -> row L%16; K-offset 0 (lanes 0-15) or 8 (lanes 16-31); two b128 loads.
__device__ __forceinline__ v16bf load_frag(const unsigned short* base, int stride, int lane) {
    int r  = lane & 15;
    int ko = (lane & 16) ? 8 : 0;
    const unsigned short* p = base + r * stride + ko;
    FragU f;
    f.q[0] = *(const uint4*)(p);
    f.q[1] = *(const uint4*)(p + 16);
    return f.v;
}

__device__ __forceinline__ v8f wmma_bf16(v16bf a, v16bf b, v8f c) {
    return __builtin_amdgcn_wmma_f32_16x16x32_bf16(false, a, false, b, (short)0, c, false, false);
}

__device__ __forceinline__ float rmax16(float v) {
#pragma unroll
    for (int m = 1; m <= 8; m <<= 1) v = fmaxf(v, __shfl_xor(v, m, 32));
    return v;
}
__device__ __forceinline__ float rsum16(float v) {
#pragma unroll
    for (int m = 1; m <= 8; m <<= 1) v += __shfl_xor(v, m, 32);
    return v;
}

// 16-byte global -> LDS copy. Async path issues GLOBAL_LOAD_ASYNC_TO_LDS_B128
// (per-lane LDS byte offset in VDST, 64-bit global address in VADDR).
__device__ __forceinline__ void copy16_to_lds(unsigned short* lds_dst, const unsigned short* gsrc) {
#if USE_ASYNC_LDS
    unsigned lds_off = (unsigned)(unsigned long long)(uintptr_t)lds_dst;  // flat->LDS: addr[31:0]
    asm volatile("global_load_async_to_lds_b128 %0, %1, off"
                 :: "v"(lds_off), "v"((unsigned long long)(uintptr_t)gsrc)
                 : "memory");
#else
    *(uint4*)lds_dst = *(const uint4*)gsrc;
#endif
}
__device__ __forceinline__ void async_fence() {
#if USE_ASYNC_LDS
    asm volatile("s_wait_asynccnt 0x0" ::: "memory");
#endif
}

// ---------------- prepass: fp32 -> bf16 convert / transpose ----------------
__global__ void k_cvt(const float* __restrict__ src, unsigned short* __restrict__ dst, int n) {
    int i = blockIdx.x * blockDim.x + threadIdx.x;
    if (i < n) dst[i] = f2bf(src[i]);
}

// src: rows x cols (row-major fp32) -> dst: cols x rows (row-major bf16)
__global__ void k_transpose(const float* __restrict__ src, unsigned short* __restrict__ dst,
                            int rows, int cols) {
    int o = blockIdx.x * blockDim.x + threadIdx.x;
    if (o >= rows * cols) return;
    int r = o % rows;       // src row
    int c = o / rows;       // src col == dst row
    dst[o] = f2bf(src[r * cols + c]);
}

// ---------------- QKV GEMM: (8192x256)@(256x6144), scatter into Q,K,V^T ----
// Each wave computes a 16x64 strip (4 tiles), reusing the A fragment 4x.
__global__ __launch_bounds__(128) void k_qkv(const unsigned short* __restrict__ xb,
                                             const unsigned short* __restrict__ wqT,
                                             unsigned short* __restrict__ Qb,
                                             unsigned short* __restrict__ Kb,
                                             unsigned short* __restrict__ VT) {
    int lane = threadIdx.x & 31;
    int wid  = blockIdx.x * 4 + (threadIdx.x >> 5);
    int tm = wid / 96, st = wid % 96;          // 512 M-tiles x 96 strips of 64 cols
    const unsigned short* abase = xb  + tm * 16 * 256;
    const unsigned short* bbase = wqT + st * 64 * 256;
    v8f acc[4];
#pragma unroll
    for (int t = 0; t < 4; ++t)
#pragma unroll
        for (int j = 0; j < 8; ++j) acc[t][j] = 0.f;
#pragma unroll
    for (int cc = 0; cc < 8; ++cc) {
        v16bf a = load_frag(abase + cc * 32, 256, lane);
#pragma unroll
        for (int t = 0; t < 4; ++t) {
            v16bf b = load_frag(bbase + t * 16 * 256 + cc * 32, 256, lane);
            acc[t] = wmma_bf16(a, b, acc[t]);
        }
    }
    int rbase = tm * 16 + ((lane & 16) ? 8 : 0);
#pragma unroll
    for (int t = 0; t < 4; ++t) {
        int j16 = st * 64 + t * 16;
        int s = j16 >> 11;                     // 0=Q 1=K 2=V (wave-uniform)
        int h = (j16 >> 8) & 7;
        int c = (j16 & 255) + (lane & 15);
#pragma unroll
        for (int j = 0; j < 8; ++j) {
            int m = rbase + j;
            int b_ = m >> 11, n = m & 2047;
            int bh = b_ * Hh + h;
            unsigned short val = f2bf(acc[t][j]);
            if (s == 0)      Qb[(bh * Nn + n) * Cc + c] = val;
            else if (s == 1) Kb[(bh * Nn + n) * Cc + c] = val;
            else             VT[(bh * Cc + c) * Nn + n] = val;   // V transposed
        }
    }
}

// ---------------- fused causal flash-attention ----------------------------
// Workgroup = 4 waves covering 64 consecutive query rows of one (b,h).
// K/V key-blocks staged in LDS once per workgroup via async global->LDS DMA.
__global__ __launch_bounds__(128) void k_attn(const unsigned short* __restrict__ Qb,
                                              const unsigned short* __restrict__ Kb,
                                              const unsigned short* __restrict__ VT,
                                              unsigned short* __restrict__ Ob) {
    __shared__ unsigned short kbuf[32 * 256];      // K rows j0..j0+31, stride 256
    __shared__ unsigned short vbuf[256 * 32];      // V^T rows c=0..255, 32 keys
    __shared__ unsigned short qbuf[4][16 * 256];   // per-wave Q tile
    __shared__ unsigned short pbuf[4][16 * 32];    // per-wave P tile

    int tid  = threadIdx.x;
    int lane = tid & 31;
    int w    = tid >> 5;
    int bh   = blockIdx.x >> 5;                    // 32 blocks of 64 rows per (b,h)
    int q0w  = (blockIdx.x & 31) * 64;
    int q0   = q0w + w * 16;
    int col0 = lane & 15;
    int rb   = (lane & 16) ? 8 : 0;

    const unsigned short* kbase = Kb + bh * Nn * Cc;
    const unsigned short* vbase = VT + bh * Cc * Nn;

    // stage this wave's Q tile (16x256 = contiguous 8KB) into LDS
    {
        const unsigned short* qsrc = Qb + (bh * Nn + q0) * Cc;
#pragma unroll
        for (int i = 0; i < 16; ++i) {
            int idx = i * 32 + lane;
            copy16_to_lds(&qbuf[w][idx * 8], qsrc + idx * 8);
        }
    }

    v8f o[16];
#pragma unroll
    for (int i = 0; i < 16; ++i)
#pragma unroll
        for (int j = 0; j < 8; ++j) o[i][j] = 0.f;
    float mrow[8], lrow[8];
#pragma unroll
    for (int j = 0; j < 8; ++j) { mrow[j] = -__builtin_inff(); lrow[j] = 0.f; }

    int nb_w   = (q0 + 47) >> 5;                   // key blocks this wave needs
    int nb_max = (q0w + 48 + 47) >> 5;             // max over the 4 waves
    for (int jb = 0; jb < nb_max; ++jb) {
        int j0 = jb * 32;
        // cooperative stage: K block is a contiguous 16KB slab
        const unsigned short* ksrc = kbase + j0 * Cc;
#pragma unroll
        for (int i = 0; i < 8; ++i) {
            int idx = i * 128 + tid;               // 1024 x 16B
            copy16_to_lds(&kbuf[idx * 8], ksrc + idx * 8);
        }
        // cooperative stage: V^T slab, 256 rows x 64B (strided in global)
        const unsigned short* vsrc = vbase + j0;
#pragma unroll
        for (int i = 0; i < 8; ++i) {
            int idx = i * 128 + tid;
            int row = idx >> 2, u = idx & 3;
            copy16_to_lds(&vbuf[row * 32 + u * 8], vsrc + row * Nn + u * 8);
        }
        async_fence();
        __syncthreads();

        if (jb < nb_w) {                           // wave-uniform predicate
            v8f s0, s1;
#pragma unroll
            for (int j = 0; j < 8; ++j) { s0[j] = 0.f; s1[j] = 0.f; }
#pragma unroll
            for (int cc = 0; cc < 8; ++cc) {
                v16bf qf = load_frag(&qbuf[w][cc * 32], 256, lane);
                v16bf k0 = load_frag(&kbuf[cc * 32], 256, lane);
                v16bf k1 = load_frag(&kbuf[16 * 256 + cc * 32], 256, lane);
                s0 = wmma_bf16(qf, k0, s0);
                s1 = wmma_bf16(qf, k1, s1);
            }
            bool needmask = (j0 + 31) > q0;        // wave-uniform
            float al[8];
#pragma unroll
            for (int j = 0; j < 8; ++j) {
                float v0 = s0[j] * SCALE, v1 = s1[j] * SCALE;
                if (needmask) {
                    int row = q0 + rb + j;
                    v0 = (j0 + col0      > row) ? -__builtin_inff() : v0;
                    v1 = (j0 + 16 + col0 > row) ? -__builtin_inff() : v1;
                }
                float mn = fmaxf(mrow[j], rmax16(fmaxf(v0, v1)));
                float a  = __expf(mrow[j] - mn);
                float p0 = __expf(v0 - mn);
                float p1 = __expf(v1 - mn);
                lrow[j] = lrow[j] * a + rsum16(p0 + p1);
                mrow[j] = mn;
                al[j] = a;
                pbuf[w][(rb + j) * 32 + col0]      = f2bf(p0);
                pbuf[w][(rb + j) * 32 + col0 + 16] = f2bf(p1);
            }
#pragma unroll
            for (int ck = 0; ck < 16; ++ck)
#pragma unroll
                for (int j = 0; j < 8; ++j) o[ck][j] *= al[j];
            // P tile back from LDS in A-fragment layout (same-wave DS ops in-order)
            v16bf pa = load_frag(&pbuf[w][0], 32, lane);
#pragma unroll
            for (int ck = 0; ck < 16; ++ck) {
                v16bf vf = load_frag(&vbuf[(ck * 16) * 32], 32, lane);
                o[ck] = wmma_bf16(pa, vf, o[ck]);
            }
        }
        __syncthreads();
    }

    int b_ = bh >> 3, h = bh & 7;
#pragma unroll
    for (int j = 0; j < 8; ++j) lrow[j] = 1.f / lrow[j];
#pragma unroll
    for (int ck = 0; ck < 16; ++ck)
#pragma unroll
        for (int j = 0; j < 8; ++j) {
            int n = q0 + rb + j;
            Ob[(b_ * Nn + n) * (Hh * Cc) + h * Cc + ck * 16 + col0] = f2bf(o[ck][j] * lrow[j]);
        }
}

// ---------------- output projection: (8192x2048)@(2048x256) + bias --------
// Each wave computes a 16x64 strip (4 tiles), reusing the A fragment 4x.
__global__ __launch_bounds__(128) void k_proj(const unsigned short* __restrict__ Ob,
                                              const unsigned short* __restrict__ wpT,
                                              const float* __restrict__ bias,
                                              float* __restrict__ out) {
    int lane = threadIdx.x & 31;
    int wid  = blockIdx.x * 4 + (threadIdx.x >> 5);
    int tm = wid >> 2, st = wid & 3;               // 512 M-tiles x 4 strips of 64 cols
    const unsigned short* abase = Ob  + tm * 16 * 2048;
    const unsigned short* bbase = wpT + st * 64 * 2048;
    v8f acc[4];
#pragma unroll
    for (int t = 0; t < 4; ++t)
#pragma unroll
        for (int j = 0; j < 8; ++j) acc[t][j] = 0.f;
#pragma unroll 4
    for (int cc = 0; cc < 64; ++cc) {
        v16bf a = load_frag(abase + cc * 32, 2048, lane);
#pragma unroll
        for (int t = 0; t < 4; ++t) {
            v16bf b = load_frag(bbase + t * 16 * 2048 + cc * 32, 2048, lane);
            acc[t] = wmma_bf16(a, b, acc[t]);
        }
    }
    int rbase = tm * 16 + ((lane & 16) ? 8 : 0);
#pragma unroll
    for (int t = 0; t < 4; ++t) {
        int c = st * 64 + t * 16 + (lane & 15);
        float bv = bias[c];
#pragma unroll
        for (int j = 0; j < 8; ++j)
            out[(rbase + j) * Cc + c] = acc[t][j] + bv;
    }
}

extern "C" void kernel_launch(void* const* d_in, const int* in_sizes, int n_in,
                              void* d_out, int out_size, void* d_ws, size_t ws_size,
                              hipStream_t stream) {
    const float* x      = (const float*)d_in[0];
    const float* w_qkv  = (const float*)d_in[1];
    const float* w_proj = (const float*)d_in[2];
    const float* b_proj = (const float*)d_in[3];
    float* out = (float*)d_out;

    unsigned short* xb  = (unsigned short*)d_ws;                 // 8192*256
    unsigned short* wqT = xb  + 8192 * 256;                      // 6144*256
    unsigned short* wpT = wqT + 6144 * 256;                      // 256*2048
    unsigned short* Qb  = wpT + 256 * 2048;                      // 32*2048*256
    unsigned short* Kb  = Qb  + Bb * Hh * Nn * Cc;
    unsigned short* VT  = Kb  + Bb * Hh * Nn * Cc;
    unsigned short* Ob  = VT  + Bb * Hh * Nn * Cc;               // 8192*2048

    k_cvt      <<<8192, 256, 0, stream>>>(x, xb, 8192 * 256);
    k_transpose<<<6144, 256, 0, stream>>>(w_qkv, wqT, 256, 6144);
    k_transpose<<<2048, 256, 0, stream>>>(w_proj, wpT, 2048, 256);
    k_qkv      <<<12288, 128, 0, stream>>>(xb, wqT, Qb, Kb, VT);
    k_attn     <<<1024, 128, 0, stream>>>(Qb, Kb, VT, Ob);
    k_proj     <<<512, 128, 0, stream>>>(Ob, wpT, b_proj, out);
}
